// LightweightCosineTransformer_1589137900146
// MI455X (gfx1250) — compile-verified, matrix operand order checked by reading, not run on hardware
//
#include <hip/hip_runtime.h>
#include <math.h>

// ---------------------------------------------------------------------------
// Types for CDNA5 WMMA (wave32): D(f32 16x16) = A(bf16 16x32) x B(bf16 32x16) + C
// ---------------------------------------------------------------------------
typedef __attribute__((ext_vector_type(8)))  float  v8f;
typedef __attribute__((ext_vector_type(8)))  __bf16 v8bf;
typedef __attribute__((ext_vector_type(16))) __bf16 v16bf;

#define WMMA_BF16(A, B, C) \
  __builtin_amdgcn_wmma_f32_16x16x32_bf16(false, (A), false, (B), (short)0, (C), false, false)

// Async global->LDS copy (CDNA5): each active lane moves 16B from
// (base SGPR-pair + 32-bit lane voffset) into LDS at the lane's 32-bit LDS
// byte address. Tracked by ASYNCcnt.  (CDNA5 ISA)
#define ASYNC_LDS_B128(ldsaddr, base, voff)                                 \
  asm volatile("global_load_async_to_lds_b128 %0, %1, %2"                   \
               :: "v"((unsigned)(ldsaddr)), "v"((unsigned)(voff)),          \
                  "s"(base)                                                 \
               : "memory")

#define WAIT_ASYNC(n) asm volatile("s_wait_asynccnt " #n ::: "memory")

// Low 32 bits of a generic __shared__ address == wave-relative LDS byte offset
// (LDS aperture keeps the offset in addr[31:0]).
#define LDS_ADDR(p) ((unsigned)(size_t)(p))

// Load one 16x32 bf16 A/B fragment for this lane.
// ISA layout (16-bit A 16x32): lanes 0-15 hold row M=lane, K in {0..7, 16..23};
// lanes 16-31 hold row M=lane-16, K in {8..15, 24..31}.
static __device__ __forceinline__ v16bf ldfrag(const __bf16* p0, const __bf16* p1) {
  v8bf lo = *(const v8bf*)p0;
  v8bf hi = *(const v8bf*)p1;
  v16bf r;
#pragma unroll
  for (int i = 0; i < 8; ++i) { r[i] = lo[i]; r[i + 8] = hi[i]; }
  return r;
}

// ---------------------------------------------------------------------------
// fp32 -> bf16 convert (weights)
// ---------------------------------------------------------------------------
__global__ void cvt_bf16_kernel(const float* __restrict__ in, __bf16* __restrict__ out, int n) {
  int i = blockIdx.x * blockDim.x + threadIdx.x;
  int stride = gridDim.x * blockDim.x;
  for (; i < n; i += stride) out[i] = (__bf16)in[i];
}

// ---------------------------------------------------------------------------
// LayerNorm over rows of length 1024 -> bf16. One block (256 thr) per row.
// ---------------------------------------------------------------------------
__global__ void __launch_bounds__(256) ln_bf16_kernel(const float* __restrict__ x,
                                                      const float* __restrict__ g,
                                                      const float* __restrict__ b,
                                                      __bf16* __restrict__ out) {
  __shared__ float red[8];
  const int row  = blockIdx.x;
  const int tid  = threadIdx.x;
  const int lane = tid & 31;
  const int wave = tid >> 5;
  const float* xr = x + (size_t)row * 1024;
  float4 v = *(const float4*)(xr + tid * 4);

  float s = v.x + v.y + v.z + v.w;
#pragma unroll
  for (int m = 1; m < 32; m <<= 1) s += __shfl_xor(s, m);
  if (lane == 0) red[wave] = s;
  __syncthreads();
  float mu = 0.f;
#pragma unroll
  for (int i = 0; i < 8; ++i) mu += red[i];
  mu *= (1.0f / 1024.0f);
  __syncthreads();

  const float dx = v.x - mu, dy = v.y - mu, dz = v.z - mu, dw = v.w - mu;
  float q = dx * dx + dy * dy + dz * dz + dw * dw;
#pragma unroll
  for (int m = 1; m < 32; m <<= 1) q += __shfl_xor(q, m);
  if (lane == 0) red[wave] = q;
  __syncthreads();
  float var = 0.f;
#pragma unroll
  for (int i = 0; i < 8; ++i) var += red[i];
  var *= (1.0f / 1024.0f);
  const float inv = rsqrtf(var + 1e-5f);

  const int c0 = tid * 4;
  __bf16* o = out + (size_t)row * 1024 + c0;
  o[0] = (__bf16)(dx * inv * g[c0 + 0] + b[c0 + 0]);
  o[1] = (__bf16)(dy * inv * g[c0 + 1] + b[c0 + 1]);
  o[2] = (__bf16)(dz * inv * g[c0 + 2] + b[c0 + 2]);
  o[3] = (__bf16)(dw * inv * g[c0 + 3] + b[c0 + 3]);
}

// ---------------------------------------------------------------------------
// Split qkv[8192,3072] -> l2norm(q)/temp, l2norm(k), v  as bf16 [b*h][n][64].
// One wave per (b,h,n) row; 2 elements per lane.
// ---------------------------------------------------------------------------
__global__ void __launch_bounds__(256) pack_qkv_kernel(const float* __restrict__ qkv,
                                                       const float* __restrict__ tptr,
                                                       __bf16* __restrict__ Qn,
                                                       __bf16* __restrict__ Kn,
                                                       __bf16* __restrict__ Vb) {
  const int lane = threadIdx.x & 31;
  const int row  = blockIdx.x * 8 + (threadIdx.x >> 5);  // (b*16+h)*2048 + n
  const int b = row >> 15;
  const int h = (row >> 11) & 15;
  const int n = row & 2047;
  const size_t src = ((size_t)(b * 2048 + n)) * 3072 + h * 64 + lane * 2;
  float2 qv = *(const float2*)(qkv + src);
  float2 kv = *(const float2*)(qkv + src + 1024);
  float2 vv = *(const float2*)(qkv + src + 2048);

  float sq = qv.x * qv.x + qv.y * qv.y;
  float sk = kv.x * kv.x + kv.y * kv.y;
#pragma unroll
  for (int m = 1; m < 32; m <<= 1) { sq += __shfl_xor(sq, m); sk += __shfl_xor(sk, m); }

  const float t  = tptr[0];
  const float qs = 1.0f / (fmaxf(sqrtf(sq), 1e-12f) * t);  // fold 1/temperature into q
  const float ks = 1.0f / fmaxf(sqrtf(sk), 1e-12f);

  const size_t dst = (size_t)row * 64 + lane * 2;
  Qn[dst] = (__bf16)(qv.x * qs); Qn[dst + 1] = (__bf16)(qv.y * qs);
  Kn[dst] = (__bf16)(kv.x * ks); Kn[dst + 1] = (__bf16)(kv.y * ks);
  Vb[dst] = (__bf16)vv.x;        Vb[dst + 1] = (__bf16)vv.y;
}

// ---------------------------------------------------------------------------
// NT GEMM: out[M,N] = A[M,K](bf16) * Bt[N,K](bf16)^T, fp32 accumulate.
// Block tile 256x128, 256 threads = 8 waves (4x2), wave tile 64x64 (16 WMMA/step).
// LDS double-buffered, filled by GLOBAL_LOAD_ASYNC_TO_LDS_B128 (ASYNCcnt).
// EPI: 0 = f32 store, 1 = f32 + residual, 2 = bf16, 3 = bf16 + relu.
// ---------------------------------------------------------------------------
template <int EPI>
__global__ void __launch_bounds__(256) gemm_nt_kernel(const __bf16* __restrict__ A,
                                                      const __bf16* __restrict__ Bt,
                                                      const float* __restrict__ resid,
                                                      float* __restrict__ outF,
                                                      __bf16* __restrict__ outH,
                                                      int M, int N, int K) {
  (void)M;
  constexpr int LDT = 40;  // 32 halves + 8 pad (80B row, 16B aligned)
  __shared__ __bf16 As[2][256 * LDT];
  __shared__ __bf16 Bs[2][128 * LDT];

  const int tid  = threadIdx.x;
  const int lane = tid & 31;
  const int wave = tid >> 5;
  const int wm = wave >> 1;   // 0..3 -> M offset wm*64
  const int wn = wave & 1;    // 0..1 -> N offset wn*64
  const int bm = blockIdx.y * 256;
  const int bn = blockIdx.x * 128;

  const __bf16* Abase = A + (size_t)bm * K;   // block-uniform -> SGPR pair
  const __bf16* Bbase = Bt + (size_t)bn * K;

  // Per-lane async mapping: A: lane owns row `tid` (32 halves = 4x16B chunks);
  // B: lane owns half of row `tid>>1` (16 halves = 2x16B chunks).
  const unsigned aOff = (unsigned)tid * (unsigned)(K * 2);
  const unsigned bOff = (unsigned)(tid >> 1) * (unsigned)(K * 2) + (unsigned)(tid & 1) * 32u;

  const unsigned aLds0 = LDS_ADDR(&As[0][tid * LDT]);
  const unsigned aLds1 = LDS_ADDR(&As[1][tid * LDT]);
  const unsigned bLds0 = LDS_ADDR(&Bs[0][(tid >> 1) * LDT + (tid & 1) * 16]);
  const unsigned bLds1 = LDS_ADDR(&Bs[1][(tid >> 1) * LDT + (tid & 1) * 16]);

#define GEMM_ISSUE(step, al, bl)                          \
  do {                                                    \
    const __bf16* ab = Abase + (size_t)(step) * 32;       \
    const __bf16* bb = Bbase + (size_t)(step) * 32;       \
    ASYNC_LDS_B128((al) + 0u,  ab, aOff + 0u);            \
    ASYNC_LDS_B128((al) + 16u, ab, aOff + 16u);           \
    ASYNC_LDS_B128((al) + 32u, ab, aOff + 32u);           \
    ASYNC_LDS_B128((al) + 48u, ab, aOff + 48u);           \
    ASYNC_LDS_B128((bl) + 0u,  bb, bOff + 0u);            \
    ASYNC_LDS_B128((bl) + 16u, bb, bOff + 16u);           \
  } while (0)

  const int ns = K / 32;
  GEMM_ISSUE(0, aLds0, bLds0);
  GEMM_ISSUE(1, aLds1, bLds1);

  v8f acc[4][4] = {};
  const int r  = lane & 15;
  const int kh = (lane >> 4) * 8;  // halves

  for (int i = 0; i < ns; ++i) {
    // In-order retirement: <=6 outstanding means the current set has landed
    // while the next set may remain in flight.
    if (i + 1 < ns) WAIT_ASYNC(6); else WAIT_ASYNC(0);
    __syncthreads();

    const __bf16* Ab = (i & 1) ? &As[1][0] : &As[0][0];
    const __bf16* Bb = (i & 1) ? &Bs[1][0] : &Bs[0][0];

    v16bf af[4], bfr[4];
#pragma unroll
    for (int u = 0; u < 4; ++u) {
      const __bf16* p = &Ab[(wm * 64 + u * 16 + r) * LDT + kh];
      af[u] = ldfrag(p, p + 16);
    }
#pragma unroll
    for (int v = 0; v < 4; ++v) {
      const __bf16* p = &Bb[(wn * 64 + v * 16 + r) * LDT + kh];
      bfr[v] = ldfrag(p, p + 16);
    }
#pragma unroll
    for (int u = 0; u < 4; ++u)
#pragma unroll
      for (int v = 0; v < 4; ++v)
        acc[u][v] = WMMA_BF16(af[u], bfr[v], acc[u][v]);

    __syncthreads();  // all waves done reading buf[i&1]
    if (i + 2 < ns) {
      if (i & 1) GEMM_ISSUE(i + 2, aLds1, bLds1);
      else       GEMM_ISSUE(i + 2, aLds0, bLds0);
    }
  }
#undef GEMM_ISSUE

  // C layout: lane col = lane&15; rows (lane>>4)*8 + rr
  const int nn = lane & 15;
  const int mo = (lane >> 4) * 8;
#pragma unroll
  for (int u = 0; u < 4; ++u)
#pragma unroll
    for (int v = 0; v < 4; ++v) {
      const int row0 = bm + wm * 64 + u * 16 + mo;
      const int col  = bn + wn * 64 + v * 16 + nn;
#pragma unroll
      for (int rr = 0; rr < 8; ++rr) {
        const size_t idx = (size_t)(row0 + rr) * N + col;
        float val = acc[u][v][rr];
        if constexpr (EPI == 0) outF[idx] = val;
        else if constexpr (EPI == 1) outF[idx] = val + resid[idx];
        else if constexpr (EPI == 2) outH[idx] = (__bf16)val;
        else outH[idx] = (__bf16)fmaxf(val, 0.0f);
      }
    }
}

// ---------------------------------------------------------------------------
// Flash-style attention. One block = one (b,h), 64 query rows; 4 waves x 16 rows.
// Qn/Kn already l2-normalized (1/temp folded into Qn). d_head = 64, N = 2048.
// K tile loaded via async global->LDS; V transposed through VGPRs.
// Output bf16 into attn_out[b*2048+n][h*64+d] ([8192,1024]).
// ---------------------------------------------------------------------------
__global__ void __launch_bounds__(128) attn_kernel(const __bf16* __restrict__ Qn,
                                                   const __bf16* __restrict__ Kn,
                                                   const __bf16* __restrict__ Vb,
                                                   __bf16* __restrict__ Ob) {
  constexpr int LDK = 72;  // 64 halves + 8 pad
  __shared__ __bf16 Ks[64 * LDK];  // [key][d]
  __shared__ __bf16 Vt[64 * LDK];  // [d][key]  (transposed V)
  __shared__ __bf16 Ps[64 * LDK];  // [qrow][key] bf16 probabilities

  const int tid  = threadIdx.x;
  const int lane = tid & 31;
  const int wave = tid >> 5;  // 0..3
  const int bh = blockIdx.y;
  const int b  = bh >> 4, h = bh & 15;
  const int q0 = blockIdx.x * 64;
  const size_t base = (size_t)bh * 2048 * 64;

  const int r  = lane & 15;
  const int kh = (lane >> 4) * 8;
  const int nn = lane & 15;
  const int mo = (lane >> 4) * 8;

  // Q fragments for this wave's 16 rows (K = 64 -> two 16x32 chunks), in regs.
  const __bf16* qrow = Qn + base + (size_t)(q0 + wave * 16 + r) * 64;
  const v16bf qf0 = ldfrag(qrow + kh, qrow + kh + 16);
  const v16bf qf1 = ldfrag(qrow + 32 + kh, qrow + 32 + kh + 16);

  v8f o[4] = {};          // 16x64 output accum (4 tiles)
  float mrow[8], lrow[8]; // per-row running max / sum (rows mo..mo+7)
#pragma unroll
  for (int i = 0; i < 8; ++i) { mrow[i] = -1e30f; lrow[i] = 0.f; }

  const int krow = tid >> 1;
  const int kc   = (tid & 1) * 32;  // halves
  const unsigned kOff  = (unsigned)(krow * 128 + kc * 2);  // bytes within K tile
  const unsigned ksLds = LDS_ADDR(&Ks[krow * LDK + kc]);

  for (int kb = 0; kb < 2048; kb += 64) {
    // V tile through VGPRs (needs transpose), K tile async -> LDS.
    const __bf16* vg = Vb + base + (size_t)(kb + krow) * 64 + kc;
    v8bf vvv[4];
#pragma unroll
    for (int c = 0; c < 4; ++c) vvv[c] = *(const v8bf*)(vg + c * 8);

    __syncthreads();  // previous iteration finished reading Ks/Vt/Ps

    const __bf16* kbase = Kn + base + (size_t)kb * 64;  // block-uniform
    ASYNC_LDS_B128(ksLds + 0u,  kbase, kOff + 0u);
    ASYNC_LDS_B128(ksLds + 16u, kbase, kOff + 16u);
    ASYNC_LDS_B128(ksLds + 32u, kbase, kOff + 32u);
    ASYNC_LDS_B128(ksLds + 48u, kbase, kOff + 48u);

#pragma unroll
    for (int c = 0; c < 4; ++c)
#pragma unroll
      for (int e = 0; e < 8; ++e) Vt[(kc + c * 8 + e) * LDK + krow] = vvv[c][e];

    WAIT_ASYNC(0);
    __syncthreads();

    // S = Q * K^T  (16 q-rows x 64 keys per wave)
    v8f s[4] = {};
#pragma unroll
    for (int j = 0; j < 4; ++j) {
      const __bf16* p = &Ks[(j * 16 + r) * LDK + kh];
      s[j] = WMMA_BF16(qf0, ldfrag(p, p + 16), s[j]);
      s[j] = WMMA_BF16(qf1, ldfrag(p + 32, p + 48), s[j]);
    }

    // online softmax: component rr = row mo+rr; its 16 cols span the lane-half
#pragma unroll
    for (int rr = 0; rr < 8; ++rr) {
      float mx = fmaxf(fmaxf(s[0][rr], s[1][rr]), fmaxf(s[2][rr], s[3][rr]));
      mx = fmaxf(mx, __shfl_xor(mx, 1));
      mx = fmaxf(mx, __shfl_xor(mx, 2));
      mx = fmaxf(mx, __shfl_xor(mx, 4));
      mx = fmaxf(mx, __shfl_xor(mx, 8));
      const float mnew  = fmaxf(mrow[rr], mx);
      const float alpha = __expf(mrow[rr] - mnew);
      float sm = 0.f;
#pragma unroll
      for (int j = 0; j < 4; ++j) {
        float pv = __expf(s[j][rr] - mnew);
        s[j][rr] = pv;
        sm += pv;
      }
      sm += __shfl_xor(sm, 1);
      sm += __shfl_xor(sm, 2);
      sm += __shfl_xor(sm, 4);
      sm += __shfl_xor(sm, 8);
      lrow[rr] = lrow[rr] * alpha + sm;
      mrow[rr] = mnew;
#pragma unroll
      for (int j = 0; j < 4; ++j) o[j][rr] *= alpha;
    }

    // C-layout -> A-layout for P via LDS
#pragma unroll
    for (int j = 0; j < 4; ++j)
#pragma unroll
      for (int rr = 0; rr < 8; ++rr)
        Ps[(wave * 16 + mo + rr) * LDK + j * 16 + nn] = (__bf16)s[j][rr];
    __syncthreads();

    // O += P * V  (A = P 16x64, B = V^T rows d-major)
    const __bf16* pp = &Ps[(wave * 16 + r) * LDK + kh];
    const v16bf pf0 = ldfrag(pp, pp + 16);
    const v16bf pf1 = ldfrag(pp + 32, pp + 48);
#pragma unroll
    for (int j = 0; j < 4; ++j) {
      const __bf16* vp = &Vt[(j * 16 + r) * LDK + kh];
      o[j] = WMMA_BF16(pf0, ldfrag(vp, vp + 16), o[j]);
      o[j] = WMMA_BF16(pf1, ldfrag(vp + 32, vp + 48), o[j]);
    }
  }

  // finalize: divide by l, write bf16 to [b*2048+n][h*64+d]
#pragma unroll
  for (int j = 0; j < 4; ++j)
#pragma unroll
    for (int rr = 0; rr < 8; ++rr) {
      const int nrow = q0 + wave * 16 + mo + rr;
      const size_t idx = (size_t)(b * 2048 + nrow) * 1024 + h * 64 + j * 16 + nn;
      Ob[idx] = (__bf16)(o[j][rr] / lrow[rr]);
    }
}

// ---------------------------------------------------------------------------
// Host orchestration
// ---------------------------------------------------------------------------
extern "C" void kernel_launch(void* const* d_in, const int* in_sizes, int n_in,
                              void* d_out, int out_size, void* d_ws, size_t ws_size,
                              hipStream_t stream) {
  (void)in_sizes; (void)n_in; (void)out_size; (void)ws_size;

  const float* x     = (const float*)d_in[0];
  const float* w_qkv = (const float*)d_in[1];
  const float* w_out = (const float*)d_in[2];
  const float* w_ff1 = (const float*)d_in[3];
  const float* w_ff2 = (const float*)d_in[4];
  const float* g1    = (const float*)d_in[5];
  const float* b1    = (const float*)d_in[6];
  const float* g2    = (const float*)d_in[7];
  const float* b2    = (const float*)d_in[8];
  const float* temp  = (const float*)d_in[9];

  char* ws = (char*)d_ws;
  size_t off = 0;
  auto take = [&](size_t bytes) {
    char* p = ws + off;
    off += (bytes + 255) & ~(size_t)255;
    return p;
  };

  __bf16* wqkv_h = (__bf16*)take((size_t)3072 * 1024 * 2);
  __bf16* wout_h = (__bf16*)take((size_t)1024 * 1024 * 2);
  __bf16* wff1_h = (__bf16*)take((size_t)2048 * 1024 * 2);
  __bf16* wff2_h = (__bf16*)take((size_t)1024 * 2048 * 2);
  __bf16* h_bf   = (__bf16*)take((size_t)8192 * 1024 * 2);  // ln1 out, reused for ln2 out
  char*   big    = take((size_t)8192 * 3072 * 4);           // qkv f32, later reused
  __bf16* qn     = (__bf16*)take((size_t)8192 * 1024 * 2);
  __bf16* kn     = (__bf16*)take((size_t)8192 * 1024 * 2);
  __bf16* vb     = (__bf16*)take((size_t)8192 * 1024 * 2);

  float*  qkvf    = (float*)big;                                         // [8192,3072] f32
  float*  x2      = (float*)big;                                         // [8192,1024] f32 (after pack)
  __bf16* a1      = (__bf16*)(big + (size_t)8192 * 1024 * 4);            // [8192,2048] bf16
  __bf16* attn_bf = (__bf16*)(big + (size_t)8192 * 1024 * 4 + (size_t)8192 * 2048 * 2);  // [8192,1024] bf16

  // weights -> bf16
  cvt_bf16_kernel<<<2048, 256, 0, stream>>>(w_qkv, wqkv_h, 3072 * 1024);
  cvt_bf16_kernel<<<2048, 256, 0, stream>>>(w_out, wout_h, 1024 * 1024);
  cvt_bf16_kernel<<<2048, 256, 0, stream>>>(w_ff1, wff1_h, 2048 * 1024);
  cvt_bf16_kernel<<<2048, 256, 0, stream>>>(w_ff2, wff2_h, 1024 * 2048);

  // LN1 -> h (bf16)
  ln_bf16_kernel<<<8192, 256, 0, stream>>>(x, g1, b1, h_bf);

  // qkv = h @ w_qkv^T  -> f32
  gemm_nt_kernel<0><<<dim3(3072 / 128, 8192 / 256), 256, 0, stream>>>(
      h_bf, wqkv_h, nullptr, qkvf, nullptr, 8192, 3072, 1024);

  // split + l2norm(q)/temp, l2norm(k), v -> bf16 [bh][n][64]
  pack_qkv_kernel<<<131072 / 8, 256, 0, stream>>>(qkvf, temp, qn, kn, vb);

  // attention -> bf16 [8192,1024]
  attn_kernel<<<dim3(2048 / 64, 64), 128, 0, stream>>>(qn, kn, vb, attn_bf);

  // x2 = x + attn @ w_out^T   (f32)
  gemm_nt_kernel<1><<<dim3(1024 / 128, 8192 / 256), 256, 0, stream>>>(
      attn_bf, wout_h, x, x2, nullptr, 8192, 1024, 1024);

  // LN2 -> h2 (bf16, reuse h_bf)
  ln_bf16_kernel<<<8192, 256, 0, stream>>>(x2, g2, b2, h_bf);

  // a1 = relu(h2 @ w_ffn1^T)  (bf16)
  gemm_nt_kernel<3><<<dim3(2048 / 128, 8192 / 256), 256, 0, stream>>>(
      h_bf, wff1_h, nullptr, nullptr, a1, 8192, 2048, 1024);

  // out = x2 + a1 @ w_ffn2^T  (f32, straight to d_out)
  gemm_nt_kernel<1><<<dim3(1024 / 128, 8192 / 256), 256, 0, stream>>>(
      a1, wff2_h, x2, (float*)d_out, nullptr, 8192, 1024, 2048);
}
